// C3k_CBSA_45389214384216
// MI455X (gfx1250) — compile-verified, matrix operand order checked by reading, not run on hardware
//
#include <hip/hip_runtime.h>
#include <hip/hip_bf16.h>
#include <math.h>
#include <stdint.h>

typedef __bf16 bf16;
typedef __attribute__((ext_vector_type(16))) __bf16 v16bf;
typedef __attribute__((ext_vector_type(8)))  __bf16 v8bf;
typedef __attribute__((ext_vector_type(4)))  __bf16 v4bf;
typedef __attribute__((ext_vector_type(8)))  float  v8f;
typedef __attribute__((ext_vector_type(4)))  unsigned int u32x4;
typedef __attribute__((ext_vector_type(8)))  int i32x8;
typedef __attribute__((ext_vector_type(4)))  int i32x4;

#define BM 64
#define BK 32
#define NTHR 256

#if defined(__HIP_DEVICE_COMPILE__) && __has_builtin(__builtin_amdgcn_tensor_load_to_lds)
#define TDM_OK 1
#else
#define TDM_OK 0
#endif

// ================= generic batched bf16 WMMA GEMM =================
// D[z][m][n] = epilogue( alpha * sum_k A[z][m][k] * B[z][k][n] )
// A: row-major, k contiguous (a_rm = row stride).
// B: BT=true : element (k,n) at n*b_rn + k ; BT=false: at k*b_rk + n.
// KCAP>0: A is a 64xKCAP panel (K==KCAP) DMA'd to LDS once via TDM.
// Batch offset: z -> (zr = z & (2^zsh-1), zq = z >> zsh); off = zr*lo + zq*hi.
struct GP {
    const bf16* A; const bf16* B;
    int  zsh;
    long a_lo, a_hi, b_lo, b_hi;
    long a_rm, b_rk, b_rn;
    int  M, N, K;
    const float* scale;
    const float* bias;
    int  do_silu;
    const float* alphap;      // per-batch alpha = alphap[zr] (nullable)
    float alpha_c;
    const float* Cadd; long c_lo, c_hi, ldc;
    int  trans_store;         // 1: out[n*ld + m] vector stores; 0: out[m*ld + n] scalar
    bf16*  obf; long obf_lo, obf_hi, ldobf;
    float* of;  long of_lo,  of_hi,  ldof;
};

#if TDM_OK
// Issue one TDM 2D tile load: 64 rows x KCAP bf16, row stride a_rm elems,
// into LDS at lds_off with 16B padding appended per row (pad-enable).
template<int KCAP>
__device__ __forceinline__ void tdm_issue(const bf16* gptr, unsigned lds_off, long a_rm) {
    constexpr int ICODE = (KCAP == 128) ? 5 : (KCAP == 256) ? 6 : 7; // log2(K)-2
    unsigned long long ga = (unsigned long long)(uintptr_t)gptr;
    u32x4 g0;
    g0[0] = 1u;                                   // count=1
    g0[1] = lds_off;                              // lds_addr
    g0[2] = (unsigned)ga;                         // global_addr[31:0]
    g0[3] = (unsigned)((ga >> 32) & 0x1FFFFFFu) | (2u << 30);  // addr[56:32] | type=2
    i32x8 g1;
    g1[0] = (1 << 16) | (1 << 20) | (ICODE << 22) | (3 << 25); // 2B elems, pad 4 dw / row
    g1[1] = (KCAP & 0xFFFF) << 16;                // tensor_dim0 lo
    g1[2] = ((KCAP >> 16) & 0xFFFF) | (64 << 16); // tensor_dim0 hi | tensor_dim1 lo
    g1[3] = (KCAP & 0xFFFF) << 16;                // tensor_dim1 hi=0 | tile_dim0
    g1[4] = 64;                                   // tile_dim1 (tile_dim2=0)
    g1[5] = (int)(unsigned)a_rm;                  // tensor_dim0_stride lo
    g1[6] = 0; g1[7] = 0;
    i32x4 z4 = {};
    i32x8 z8 = {};
    __builtin_amdgcn_tensor_load_to_lds(g0, g1, z4, z4, z8, 0);
    __builtin_amdgcn_s_wait_tensorcnt(0);
}
#endif

template<int TBN, bool BT, int KCAP>
__global__ __launch_bounds__(NTHR) void k_gemm(GP p) {
    constexpr int NT  = TBN / 64;
    constexpr int NBT = TBN / 16;
    constexpr int CPR = TBN / 8;
    constexpr int NCH = (BK * TBN) / (8 * NTHR);

    const int tid  = threadIdx.x;
    const int lane = tid & 31;
    const int wave = tid >> 5;
    const int wm = wave >> 2;
    const int wn = wave & 3;
    const int z  = blockIdx.z;
    const int zr = z & ((1 << p.zsh) - 1);
    const int zq = z >> p.zsh;
    const int m0 = blockIdx.y * BM;
    const int n0 = blockIdx.x * TBN;
    const bf16* A = p.A + (long)zr * p.a_lo + (long)zq * p.a_hi;
    const bf16* B = p.B + (long)zr * p.b_lo + (long)zq * p.b_hi;

    __shared__ bf16 BF[NBT][32][16];   // B tiles in WMMA fragment order

    // ---- B staging ----
    const bf16* pb[NCH];
    bf16* sB[NCH];
    #pragma unroll
    for (int c = 0; c < NCH; ++c) {
        int idx = tid + c * NTHR;
        if (BT) {
            int bn = idx >> 2, bkc = (idx & 3) * 8;
            pb[c] = B + (long)(n0 + bn) * p.b_rn + bkc;
            sB[c] = &BF[bn >> 4][(bn & 15) + 16 * ((bkc >> 3) & 1)][8 * (bkc >> 4)];
        } else {
            int bkk = idx / CPR, bnc = (idx % CPR) * 8;
            pb[c] = B + (long)bkk * p.b_rk + n0 + bnc;
            sB[c] = &BF[bnc >> 4][(bnc & 15) + 16 * ((bkk >> 3) & 1)][(bkk & 7) + 8 * ((bkk >> 4) & 1)];
        }
    }
    const long bstep = BT ? BK : BK * p.b_rk;

    const v16bf* fB[NT];
    #pragma unroll
    for (int ni = 0; ni < NT; ++ni) fB[ni] = reinterpret_cast<const v16bf*>(&BF[wn * NT + ni][lane][0]);

    v8bf rb[NCH];
    #pragma unroll
    for (int c = 0; c < NCH; ++c) { rb[c] = *reinterpret_cast<const v8bf*>(pb[c]); pb[c] += bstep; }

    v8f acc[2][NT] = {};
    const int nk = p.K / BK;
    const int kb = (lane >> 4) * 8;
    const int rl = lane & 15;

    if constexpr (KCAP > 0) {
        // ============ TDM path: whole 64xK A panel lives in LDS ============
        __shared__ bf16 Apan[BM][KCAP + 8];          // +8: row padding (TDM pad-enable)
        const bf16* gA = A + (long)m0 * p.a_rm;
#if TDM_OK
        if (wave == 0) tdm_issue<KCAP>(gA, (unsigned)(uintptr_t)&Apan[0][0], p.a_rm);
#else
        constexpr int CW = KCAP / 8;                  // chunks per row
        for (int c = tid; c < BM * CW; c += NTHR) {
            int row = c / CW, kc = (c % CW) * 8;
            *reinterpret_cast<v8bf*>(&Apan[row][kc]) =
                *reinterpret_cast<const v8bf*>(gA + (long)row * p.a_rm + kc);
        }
#endif
        __syncthreads();

        const bf16* ar[2];
        #pragma unroll
        for (int mi = 0; mi < 2; ++mi) ar[mi] = &Apan[(wm * 2 + mi) * 16 + rl][kb];

        for (int t = 0; t < nk; ++t) {
            #pragma unroll
            for (int c = 0; c < NCH; ++c) {
                if (BT) {
                    *reinterpret_cast<v8bf*>(sB[c]) = rb[c];
                } else {
                    #pragma unroll
                    for (int j = 0; j < 8; ++j) sB[c][j * 16] = rb[c][j];
                }
            }
            __syncthreads();
            if (t + 1 < nk) {
                #pragma unroll
                for (int c = 0; c < NCH; ++c) { rb[c] = *reinterpret_cast<const v8bf*>(pb[c]); pb[c] += bstep; }
            }
            v16bf afr[2], bfr[NT];
            #pragma unroll
            for (int mi = 0; mi < 2; ++mi) {
                v8bf lo = *reinterpret_cast<const v8bf*>(ar[mi] + t * BK);
                v8bf hi = *reinterpret_cast<const v8bf*>(ar[mi] + t * BK + 16);
                afr[mi] = __builtin_shufflevector(lo, hi, 0,1,2,3,4,5,6,7,8,9,10,11,12,13,14,15);
            }
            #pragma unroll
            for (int ni = 0; ni < NT; ++ni) bfr[ni] = *fB[ni];
            #pragma unroll
            for (int mi = 0; mi < 2; ++mi)
                #pragma unroll
                for (int ni = 0; ni < NT; ++ni)
                    acc[mi][ni] = __builtin_amdgcn_wmma_f32_16x16x32_bf16(
                        false, afr[mi], false, bfr[ni], (short)0, acc[mi][ni], false, false);
            __syncthreads();
        }
    } else {
        // ============ staged-A path (fragment-order LDS) ============
        __shared__ bf16 AF[4][32][16];
        const int am  = tid >> 2;
        const int akc = (tid & 3) * 8;
        const bf16* pa = A + (long)(m0 + am) * p.a_rm + akc;
        bf16* sA = &AF[am >> 4][(am & 15) + 16 * ((akc >> 3) & 1)][8 * (akc >> 4)];
        const v16bf* fA[2];
        #pragma unroll
        for (int mi = 0; mi < 2; ++mi) fA[mi] = reinterpret_cast<const v16bf*>(&AF[wm * 2 + mi][lane][0]);

        v8bf ra = *reinterpret_cast<const v8bf*>(pa); pa += BK;

        for (int t = 0; t < nk; ++t) {
            *reinterpret_cast<v8bf*>(sA) = ra;
            #pragma unroll
            for (int c = 0; c < NCH; ++c) {
                if (BT) {
                    *reinterpret_cast<v8bf*>(sB[c]) = rb[c];
                } else {
                    #pragma unroll
                    for (int j = 0; j < 8; ++j) sB[c][j * 16] = rb[c][j];
                }
            }
            __syncthreads();
            if (t + 1 < nk) {
                ra = *reinterpret_cast<const v8bf*>(pa); pa += BK;
                #pragma unroll
                for (int c = 0; c < NCH; ++c) { rb[c] = *reinterpret_cast<const v8bf*>(pb[c]); pb[c] += bstep; }
            }
            v16bf afr[2], bfr[NT];
            #pragma unroll
            for (int mi = 0; mi < 2; ++mi) afr[mi] = *fA[mi];
            #pragma unroll
            for (int ni = 0; ni < NT; ++ni) bfr[ni] = *fB[ni];
            #pragma unroll
            for (int mi = 0; mi < 2; ++mi)
                #pragma unroll
                for (int ni = 0; ni < NT; ++ni)
                    acc[mi][ni] = __builtin_amdgcn_wmma_f32_16x16x32_bf16(
                        false, afr[mi], false, bfr[ni], (short)0, acc[mi][ni], false, false);
            __syncthreads();
        }
    }

    // ---- epilogue ----
    const float alpha = p.alphap ? p.alphap[zr] : p.alpha_c;
    const long obf_off = (long)zr * p.obf_lo + (long)zq * p.obf_hi;
    const long of_off  = (long)zr * p.of_lo  + (long)zq * p.of_hi;
    const long c_off   = (long)zr * p.c_lo   + (long)zq * p.c_hi;
    const int mrb  = 8 * (lane >> 4);
    const int ncol = lane & 15;

    #pragma unroll
    for (int mi = 0; mi < 2; ++mi)
        #pragma unroll
        for (int ni = 0; ni < NT; ++ni) {
            const int mgb = m0 + (wm * 2 + mi) * 16 + mrb;
            const int ng  = n0 + (wn * NT + ni) * 16 + ncol;
            float vv[8];
            #pragma unroll
            for (int r = 0; r < 8; ++r) {
                float v = acc[mi][ni][r] * alpha;
                int mg = mgb + r;
                if (p.scale) v *= p.scale[mg];
                if (p.bias)  v += p.bias[mg];
                if (p.Cadd)  v += p.Cadd[c_off + (long)mg * p.ldc + ng];
                if (p.do_silu) v = v / (1.f + __expf(-v));
                vv[r] = v;
            }
            if (p.trans_store) {
                if (p.obf) {
                    v8bf ob;
                    #pragma unroll
                    for (int r = 0; r < 8; ++r) ob[r] = (bf16)vv[r];
                    *reinterpret_cast<v8bf*>(&p.obf[obf_off + (long)ng * p.ldobf + mgb]) = ob;
                }
                if (p.of) {
                    float* dst = &p.of[of_off + (long)ng * p.ldof + mgb];
                    *reinterpret_cast<float4*>(dst)     = make_float4(vv[0], vv[1], vv[2], vv[3]);
                    *reinterpret_cast<float4*>(dst + 4) = make_float4(vv[4], vv[5], vv[6], vv[7]);
                }
            } else {
                #pragma unroll
                for (int r = 0; r < 8; ++r) {
                    int mg = mgb + r;
                    if (p.obf) p.obf[obf_off + (long)mg * p.ldobf + ng] = (bf16)vv[r];
                    if (p.of)  p.of [of_off  + (long)mg * p.ldof  + ng] = vv[r];
                }
            }
        }
}

// ============ transpose-convert: x (b, 256c, 6400n) f32 -> Xt (b, n, c) bf16 ============
__global__ __launch_bounds__(NTHR) void k_tc(const float* __restrict__ x, bf16* __restrict__ Xt) {
    __shared__ float T[32][33];
    const int b  = blockIdx.z;
    const int c0 = blockIdx.y * 32;
    const int n0 = blockIdx.x * 32;
    const int t  = threadIdx.x;
    const int row = t >> 3, q4 = (t & 7) * 4;
    float4 f = *reinterpret_cast<const float4*>(x + ((long)b * 256 + c0 + row) * 6400 + n0 + q4);
    T[row][q4] = f.x; T[row][q4 + 1] = f.y; T[row][q4 + 2] = f.z; T[row][q4 + 3] = f.w;
    __syncthreads();
    v4bf o;
    #pragma unroll
    for (int j = 0; j < 4; ++j) o[j] = (bf16)T[q4 + j][row];
    *reinterpret_cast<v4bf*>(Xt + ((long)b * 6400 + n0 + row) * 256 + c0 + q4) = o;
}

__global__ void k_cvt(const float4* __restrict__ in, v4bf* __restrict__ out, long n4) {
    for (long i = (long)blockIdx.x * NTHR + threadIdx.x; i < n4; i += (long)gridDim.x * NTHR) {
        float4 f = in[i];
        v4bf o; o[0] = (bf16)f.x; o[1] = (bf16)f.y; o[2] = (bf16)f.z; o[3] = (bf16)f.w;
        out[i] = o;
    }
}

// ===== pooling: Pjt (b, 6400n, 512i) -> rep (b,h,64k,64d) =====
__global__ __launch_bounds__(NTHR) void k_pool(const bf16* __restrict__ Pjt,
                                               float* __restrict__ repf,
                                               bf16*  __restrict__ repb) {
    const int b = blockIdx.y;
    const int k = blockIdx.x;
    const int ky = k >> 3, kx = k & 7;
    const int t = threadIdx.x;
    const int c8 = t & 63;
    const int g  = t >> 6;
    const bf16* base = Pjt + (long)b * 6400 * 512;
    float s[8] = {};
    for (int r = 0; r < 25; ++r) {
        int idx = g * 25 + r;
        int yy = idx / 10, xx = idx - yy * 10;
        int n = (ky * 10 + yy) * 80 + kx * 10 + xx;
        v8bf v = *reinterpret_cast<const v8bf*>(base + (long)n * 512 + c8 * 8);
        #pragma unroll
        for (int j = 0; j < 8; ++j) s[j] += (float)v[j];
    }
    __shared__ float P[4][64][8];
    #pragma unroll
    for (int j = 0; j < 8; ++j) P[g][c8][j] = s[j];
    __syncthreads();
    if (t < 64) {
        const int h = t >> 3;
        const int dbase = (t & 7) * 8;
        long o = (((long)b * 8 + h) * 64 + k) * 64 + dbase;
        #pragma unroll
        for (int j = 0; j < 8; ++j) {
            float tot = (P[0][t][j] + P[1][t][j] + P[2][t][j] + P[3][t][j]) * 0.01f;
            repf[o + j] = tot;
            repb[o + j] = (bf16)tot;
        }
    }
}

// ---------- row softmax over 6400 tokens ----------
__global__ __launch_bounds__(NTHR) void k_softmax(bf16* __restrict__ attn, float sc) {
    const long row = blockIdx.x;
    v8bf* p8 = reinterpret_cast<v8bf*>(attn + row * 6400);
    __shared__ float red[NTHR];
    float mx = -1e30f;
    for (int c = threadIdx.x; c < 800; c += NTHR) {
        v8bf v = p8[c];
        #pragma unroll
        for (int i = 0; i < 8; ++i) mx = fmaxf(mx, (float)v[i] * sc);
    }
    red[threadIdx.x] = mx; __syncthreads();
    for (int s = NTHR / 2; s > 0; s >>= 1) {
        if (threadIdx.x < s) red[threadIdx.x] = fmaxf(red[threadIdx.x], red[threadIdx.x + s]);
        __syncthreads();
    }
    mx = red[0]; __syncthreads();
    float sum = 0.f;
    for (int c = threadIdx.x; c < 800; c += NTHR) {
        v8bf v = p8[c];
        #pragma unroll
        for (int i = 0; i < 8; ++i) sum += __expf((float)v[i] * sc - mx);
    }
    red[threadIdx.x] = sum; __syncthreads();
    for (int s = NTHR / 2; s > 0; s >>= 1) {
        if (threadIdx.x < s) red[threadIdx.x] += red[threadIdx.x + s];
        __syncthreads();
    }
    const float inv = 1.f / red[0];
    for (int c = threadIdx.x; c < 800; c += NTHR) {
        v8bf v = p8[c];
        #pragma unroll
        for (int i = 0; i < 8; ++i) v[i] = (bf16)(__expf((float)v[i] * sc - mx) * inv);
        p8[c] = v;
    }
}

// ---------- tiny landmark self-attention per (b,h) ----------
__global__ __launch_bounds__(NTHR) void k_lmattn(const float* __restrict__ repN,
                                                 const float* __restrict__ step_x,
                                                 bf16* __restrict__ xd2t, float sc) {
    const long z = blockIdx.x;
    const int h = (int)(z & 7);
    __shared__ float R[64][64];
    __shared__ float S2[64][65];
    const float4* r4 = reinterpret_cast<const float4*>(repN + z * 4096);
    float4* R4 = reinterpret_cast<float4*>(&R[0][0]);
    for (int e = threadIdx.x; e < 1024; e += NTHR) R4[e] = r4[e];
    __syncthreads();
    for (int e = threadIdx.x; e < 4096; e += NTHR) {
        int k = e >> 6, j = e & 63;
        float s = 0.f;
        #pragma unroll 8
        for (int d = 0; d < 64; ++d) s += R[k][d] * R[j][d];
        S2[k][j] = s * sc;
    }
    __syncthreads();
    if (threadIdx.x < 64) {
        int k = threadIdx.x;
        float mx = -1e30f;
        for (int j = 0; j < 64; ++j) mx = fmaxf(mx, S2[k][j]);
        float sum = 0.f;
        for (int j = 0; j < 64; ++j) { float e = __expf(S2[k][j] - mx); S2[k][j] = e; sum += e; }
        float inv = 1.f / sum;
        for (int j = 0; j < 64; ++j) S2[k][j] *= inv;
    }
    __syncthreads();
    const float stx = step_x[h];
    for (int e = threadIdx.x; e < 4096; e += NTHR) {
        int k = e >> 6, d = e & 63;
        float s = 0.f;
        #pragma unroll 8
        for (int j = 0; j < 64; ++j) s += S2[k][j] * R[j][d];
        xd2t[z * 4096 + (long)d * 64 + k] = (bf16)(s * stx);
    }
}

// ================= host =================
static inline size_t alignup(size_t v) { return (v + 255) & ~(size_t)255; }

extern "C" void kernel_launch(void* const* d_in, const int* in_sizes, int n_in,
                              void* d_out, int out_size, void* d_ws, size_t ws_size,
                              hipStream_t stream) {
    (void)in_sizes; (void)n_in; (void)out_size; (void)ws_size;
    const int Bb = 8, C1 = 256, C_ = 128, C2 = 256, INNER = 512, Hh = 8;
    const long Nn = 6400;
    const float SCALE = 0.125f;

    const float* x     = (const float*)d_in[0];
    const float* cv1_w = (const float*)d_in[1];
    const float* cv1_s = (const float*)d_in[2];
    const float* cv1_b = (const float*)d_in[3];
    const float* cv2_w = (const float*)d_in[4];
    const float* cv2_s = (const float*)d_in[5];
    const float* cv2_b = (const float*)d_in[6];
    const float* cv3_w = (const float*)d_in[7];
    const float* cv3_s = (const float*)d_in[8];
    const float* cv3_b = (const float*)d_in[9];
    const float* proj_w = (const float*)d_in[10];
    const float* step_rep = (const float*)d_in[11];
    const float* step_x   = (const float*)d_in[12];
    const float* out_w = (const float*)d_in[13];
    const float* out_b = (const float*)d_in[14];
    float* outp = (float*)d_out;

    char* w = (char*)d_ws;
    size_t off = 0;
    auto take = [&](size_t bytes) { char* p = w + off; off += alignup(bytes); return p; };
    bf16* Wcv1 = (bf16*)take((size_t)C_ * C1 * 2);
    bf16* Wcv2 = (bf16*)take((size_t)C_ * C1 * 2);
    bf16* Wcv3 = (bf16*)take((size_t)C2 * 2 * C_ * 2);
    bf16* Wprj = (bf16*)take((size_t)INNER * C_ * 2);
    bf16* Wout = (bf16*)take((size_t)C_ * INNER * 2);
    bf16* Xt   = (bf16*)take((size_t)Bb * Nn * C1 * 2);
    bf16* Y1t  = (bf16*)take((size_t)Bb * Nn * C_ * 2);
    bf16* Yt   = (bf16*)take((size_t)Bb * Nn * 2 * C_ * 2);
    bf16* Pjt  = (bf16*)take((size_t)Bb * Nn * INNER * 2);
    float* RepF = (float*)take((size_t)Bb * Hh * 64 * 64 * 4);
    bf16*  RepB = (bf16*) take((size_t)Bb * Hh * 64 * 64 * 2);
    float* RepNF = (float*)take((size_t)Bb * Hh * 64 * 64 * 4);
    bf16*  RepNB = (bf16*) take((size_t)Bb * Hh * 64 * 64 * 2);
    bf16* Attn = (bf16*)take((size_t)Bb * Hh * 64 * Nn * 2);
    bf16* XD2T = (bf16*)take((size_t)Bb * Hh * 64 * 64 * 2);
    bf16* XDt  = (bf16*)take((size_t)Bb * Nn * INNER * 2);

    auto cvt = [&](const float* in, bf16* out, long n) {
        long n4 = n / 4;
        long g = (n4 + NTHR - 1) / NTHR; if (g > 4096) g = 4096;
        k_cvt<<<dim3((unsigned)g), NTHR, 0, stream>>>((const float4*)in, (v4bf*)out, n4);
    };
    cvt(cv1_w, Wcv1, (long)C_ * C1);
    cvt(cv2_w, Wcv2, (long)C_ * C1);
    cvt(cv3_w, Wcv3, (long)C2 * 2 * C_);
    cvt(proj_w, Wprj, (long)INNER * C_);
    cvt(out_w, Wout, (long)C_ * INNER);
    k_tc<<<dim3(200, 8, Bb), NTHR, 0, stream>>>(x, Xt);

    const long ZS = 64L * Nn;
    const long PT = Nn * INNER;
    GP p{};

    // cv1 (BT, TDM K=256): SiLU -> Y1t
    p = GP{}; p.A = Wcv1; p.B = Xt; p.zsh = 0;
    p.b_hi = Nn * C1; p.a_rm = C1; p.b_rn = C1;
    p.M = C_; p.N = (int)Nn; p.K = C1;
    p.scale = cv1_s; p.bias = cv1_b; p.do_silu = 1; p.alpha_c = 1.f;
    p.trans_store = 1; p.obf = Y1t; p.obf_hi = Nn * C_; p.ldobf = C_;
    { dim3 g((int)Nn / 128, C_ / BM, Bb); k_gemm<128, true, 256><<<g, NTHR, 0, stream>>>(p); }

    // cv2 (BT, TDM K=256): -> Yt[:,128..255]
    p.A = Wcv2; p.scale = cv2_s; p.bias = cv2_b;
    p.obf = Yt + C_; p.obf_hi = Nn * 2 * C_; p.ldobf = 2 * C_;
    { dim3 g((int)Nn / 128, C_ / BM, Bb); k_gemm<128, true, 256><<<g, NTHR, 0, stream>>>(p); }

    // proj (BT, TDM K=128): -> Pjt
    p = GP{}; p.A = Wprj; p.B = Y1t; p.zsh = 0;
    p.b_hi = Nn * C_; p.a_rm = C_; p.b_rn = C_;
    p.M = INNER; p.N = (int)Nn; p.K = C_; p.alpha_c = 1.f;
    p.trans_store = 1; p.obf = Pjt; p.obf_hi = PT; p.ldobf = INNER;
    { dim3 g((int)Nn / 128, INNER / BM, Bb); k_gemm<128, true, 128><<<g, NTHR, 0, stream>>>(p); }

    // pooling -> rep
    k_pool<<<dim3(64, Bb), NTHR, 0, stream>>>(Pjt, RepF, RepB);

    // scores^T (BT): M=6400(tok),N=64(lm),K=64(d) -> Attn[z][lm][tok]
    p = GP{}; p.A = Pjt; p.B = RepB; p.zsh = 3;
    p.a_lo = 64; p.a_hi = PT; p.b_lo = 64L * 64; p.b_hi = 8L * 64 * 64;
    p.a_rm = INNER; p.b_rn = 64;
    p.M = (int)Nn; p.N = 64; p.K = 64; p.alpha_c = 1.f;
    p.trans_store = 1; p.obf = Attn; p.obf_lo = ZS; p.obf_hi = 8L * ZS; p.ldobf = Nn;
    { dim3 g(1, (int)Nn / BM, Bb * Hh); k_gemm<64, true, 0><<<g, NTHR, 0, stream>>>(p); }

    // softmax rows (applies SCALE)
    k_softmax<<<dim3(Bb * Hh * 64), NTHR, 0, stream>>>(Attn, SCALE);

    // rep_delta (!BT): M=64,N=64,K=6400; repN = rep + step_rep[h]*delta
    p = GP{}; p.A = Attn; p.B = Pjt; p.zsh = 3;
    p.a_lo = ZS; p.a_hi = 8L * ZS; p.b_lo = 64; p.b_hi = PT;
    p.a_rm = Nn; p.b_rk = INNER;
    p.M = 64; p.N = 64; p.K = (int)Nn;
    p.alphap = step_rep;
    p.Cadd = RepF; p.c_lo = 64L * 64; p.c_hi = 8L * 64 * 64; p.ldc = 64;
    p.trans_store = 0;
    p.obf = RepNB; p.obf_lo = 64L * 64; p.obf_hi = 8L * 64 * 64; p.ldobf = 64;
    p.of  = RepNF; p.of_lo  = 64L * 64; p.of_hi  = 8L * 64 * 64; p.ldof  = 64;
    { dim3 g(1, 1, Bb * Hh); k_gemm<64, false, 0><<<g, NTHR, 0, stream>>>(p); }

    // landmark self-attn -> XD2T
    k_lmattn<<<dim3(Bb * Hh), NTHR, 0, stream>>>(RepNF, step_x, XD2T, SCALE);

    // scatter (!BT): M=64(d),N=6400,K=64 -> XDt[b][tok][h*64+d]
    p = GP{}; p.A = XD2T; p.B = Attn; p.zsh = 3;
    p.a_lo = 64L * 64; p.a_hi = 8L * 64 * 64; p.b_lo = ZS; p.b_hi = 8L * ZS;
    p.a_rm = 64; p.b_rk = Nn;
    p.M = 64; p.N = (int)Nn; p.K = 64; p.alpha_c = 1.f;
    p.trans_store = 1; p.obf = XDt; p.obf_lo = 64; p.obf_hi = PT; p.ldobf = INNER;
    { dim3 g((int)Nn / 128, 1, Bb * Hh); k_gemm<128, false, 0><<<g, NTHR, 0, stream>>>(p); }

    // out proj (BT, staged A: K=512 panel too large for TDM LDS budget)
    p = GP{}; p.A = Wout; p.B = XDt; p.zsh = 0;
    p.b_hi = PT; p.a_rm = INNER; p.b_rn = INNER;
    p.M = C_; p.N = (int)Nn; p.K = INNER;
    p.bias = out_b; p.alpha_c = 1.f;
    p.trans_store = 1; p.obf = Yt; p.obf_hi = Nn * 2 * C_; p.ldobf = 2 * C_;
    { dim3 g((int)Nn / 128, C_ / BM, Bb); k_gemm<128, true, 0><<<g, NTHR, 0, stream>>>(p); }

    // cv3 (BT, TDM K=256): SiLU -> d_out f32 (c-major)
    p = GP{}; p.A = Wcv3; p.B = Yt; p.zsh = 0;
    p.b_hi = Nn * 2 * C_; p.a_rm = 2 * C_; p.b_rn = 2 * C_;
    p.M = C2; p.N = (int)Nn; p.K = 2 * C_;
    p.scale = cv3_s; p.bias = cv3_b; p.do_silu = 1; p.alpha_c = 1.f;
    p.trans_store = 0; p.of = outp; p.of_hi = (long)C2 * Nn; p.ldof = Nn;
    { dim3 g((int)Nn / 128, C2 / BM, Bb); k_gemm<128, true, 256><<<g, NTHR, 0, stream>>>(p); }
}
